// StackRNN_3856880632251
// MI455X (gfx1250) — matrix operand order repaired and checked by reading, not compiled, not run on hardware
//
#include <hip/hip_runtime.h>
#include <hip/hip_bf16.h>
#include <stddef.h>
#include <stdint.h>

// ---------------------------------------------------------------------------
// StackRNN persistent kernel for MI455X (gfx1250, wave32, WMMA).
//
// The recurrence is strictly sequential (T=4096), so one persistent kernel
// runs the whole sequence with 2 device-wide barriers per step. The dominant
// 2048x2048 hid2hid matvec is sharded over 128 WGs (16 rows each) and done
// with v_wmma_f32_16x16x32_f16 via a replicated-B matvec (all 16 B columns
// carry h, so D rows carry the dot products). Weights are pre-packed to f16
// in 16x32 A-fragment tiles with contiguous k order (matching B); each WG's 64KB tile is
// staged ONCE into LDS with global_load_async_to_lds_b128 (ASYNCcnt) and all
// 4096 steps then read fragments from LDS. Everything else is L2-resident.
// ---------------------------------------------------------------------------

#define T_STEPS 4096
#define NHID    2048
#define NCHAR   256
#define NSTACK  32
#define STACKD  512
#define DEPTH   4

#define NWG_H   128                  // hidden-row WGs (16 rows each)
#define NWG_O   16                   // output-head WGs (16 rows each)
#define WG_ACT  (NWG_H + NWG_O)      // 144: acts + top_val + stack update
#define WG_SMX  (WG_ACT + 1)         // 145: softmax of previous step's logits
#define NWG_TOT (WG_SMX + 1)         // 146

#define TILE_HALFS (64 * 32 * 16)    // one 16-row tile: 32768 f16 = 64KB

typedef _Float16 f16_t;
typedef __attribute__((ext_vector_type(16))) _Float16 v16h;
typedef __attribute__((ext_vector_type(8)))  float    v8f;

// ---------------------------------------------------------------------------
// Device helpers
// ---------------------------------------------------------------------------

__device__ __forceinline__ void grid_sync(unsigned* cnt, unsigned& epoch) {
  __threadfence();                       // release stores to device scope
  __syncthreads();
  ++epoch;
  if (threadIdx.x == 0) {
    atomicAdd(cnt, 1u);
    const unsigned target = epoch * (unsigned)NWG_TOT;
    while (__hip_atomic_load(cnt, __ATOMIC_ACQUIRE, __HIP_MEMORY_SCOPE_AGENT) < target) {
      __builtin_amdgcn_s_sleep(1);
    }
  }
  __syncthreads();
  __threadfence();                       // acquire: drop stale WGP-level lines
}

// Stage one 64KB packed weight tile global -> LDS with the CDNA5 async DMA
// path (ASYNCcnt-tracked), then wait and make it visible to all waves.
__device__ __forceinline__ void stage_tile_async(const f16_t* __restrict__ src,
                                                 f16_t* wlds) {
  const unsigned ldsbase = (unsigned)(uintptr_t)&wlds[0];
  const char* gsrc = (const char*)src;
#pragma unroll
  for (int i = threadIdx.x; i < TILE_HALFS * 2 / 16; i += 256) {   // 4096 x b128
    asm volatile("global_load_async_to_lds_b128 %0, %1, off"
                 :: "v"(ldsbase + (unsigned)i * 16u),
                    "v"(gsrc + (size_t)i * 16)
                 : "memory");
  }
  asm volatile("s_wait_asynccnt 0x0" ::: "memory");
  __syncthreads();
}

// Rows of the LDS-resident tile times h16 (2048 f16 in global/L2) -> out16.
// 8 waves split the 64 k-tiles (8 each); partials reduced through LDS.
__device__ __forceinline__ void wg_matvec16_lds(const f16_t* wlds,
                                                const f16_t* __restrict__ h16,
                                                float* red /*LDS[128]*/,
                                                float* out16 /*LDS[16]*/) {
  const int tid  = threadIdx.x;
  const int lane = tid & 31;
  const int wid  = tid >> 5;

  v8f c = {0.f, 0.f, 0.f, 0.f, 0.f, 0.f, 0.f, 0.f};
#pragma unroll
  for (int kk = 0; kk < 8; ++kk) {
    const int kt = wid * 8 + kk;
    v16h a = *(const v16h*)(wlds + (kt * 32 + lane) * 16);          // ds loads
    v16h b = *(const v16h*)(h16 + kt * 32 + ((lane >> 4) << 4));    // replicated B
    c = __builtin_amdgcn_wmma_f32_16x16x32_f16(false, a, false, b,
                                               (short)0, c, false, false);
  }
  if (lane == 0 || lane == 16) {
    const int half = lane >> 4;
#pragma unroll
    for (int v = 0; v < 8; ++v) red[wid * 16 + half * 8 + v] = c[v];
  }
  __syncthreads();
  if (tid < 16) {
    float s = 0.f;
#pragma unroll
    for (int w = 0; w < 8; ++w) s += red[w * 16 + tid];
    out16[tid] = s;
  }
  __syncthreads();
}

__device__ __forceinline__ void softmax256(const float* __restrict__ logits,
                                           float* __restrict__ dst,
                                           float* smx /*LDS[256]*/) {
  const int tid = threadIdx.x;
  const float v = logits[tid];
  smx[tid] = v;
  __syncthreads();
  for (int s2 = 128; s2 > 0; s2 >>= 1) {
    if (tid < s2) smx[tid] = fmaxf(smx[tid], smx[tid + s2]);
    __syncthreads();
  }
  const float m = smx[0];
  __syncthreads();
  const float e = __expf(v - m);
  smx[tid] = e;
  __syncthreads();
  for (int s2 = 128; s2 > 0; s2 >>= 1) {
    if (tid < s2) smx[tid] += smx[tid + s2];
    __syncthreads();
  }
  dst[tid] = e / smx[0];
  __syncthreads();
}

// ---------------------------------------------------------------------------
// Pre-pack kernels
// ---------------------------------------------------------------------------

// Pack f32 row-major [rows x 2048] into f16 WMMA A-fragment tiles:
// dst[((rt*64+kt)*32+lane)*16 + j] = src[row, K]
//   row = rt*16 + (lane&15)
//   K   = kt*32 + 16*(lane>>4) + j     (contiguous k order, matches B)
__global__ void pack_wmma_tiles(const float* __restrict__ src,
                                f16_t* __restrict__ dst,
                                int row_tiles) {
  const int idx = blockIdx.x * 256 + threadIdx.x;
  const int total = row_tiles * TILE_HALFS;
  if (idx >= total) return;
  const int j    = idx & 15;
  const int lane = (idx >> 4) & 31;
  const int kt   = (idx >> 9) & 63;
  const int rt   = idx >> 15;
  const int row  = rt * 16 + (lane & 15);
  const int K    = kt * 32 + ((lane >> 4) << 4) + j;
  dst[idx] = (f16_t)src[(size_t)row * NHID + K];
}

// Head matrix (128 rows): rows 0..95 = hid2act_w flattened (32,3,2048);
// rows 96..127 = hid2stack_w[s, TOP=0, :].
__global__ void pack_head_tiles(const float* __restrict__ act_w,
                                const float* __restrict__ stack_w,
                                f16_t* __restrict__ dst) {
  const int idx = blockIdx.x * 256 + threadIdx.x;
  if (idx >= 8 * TILE_HALFS) return;
  const int j    = idx & 15;
  const int lane = (idx >> 4) & 31;
  const int kt   = (idx >> 9) & 63;
  const int rt   = idx >> 15;
  const int row  = rt * 16 + (lane & 15);
  const int K    = kt * 32 + ((lane >> 4) << 4) + j;
  const float v = (row < 96)
      ? act_w[(size_t)row * NHID + K]
      : stack_w[(size_t)(row - 96) * STACKD * NHID + K];
  dst[idx] = (f16_t)v;
}

// Dense slice of stack2hid: Spack[n][s*4+d] = stack2hid_w[s][n][d], d<DEPTH.
__global__ void pack_spack(const float* __restrict__ s2h,
                           float* __restrict__ dst) {
  const int idx = blockIdx.x * 256 + threadIdx.x;
  if (idx >= NHID * NSTACK * DEPTH) return;
  const int n  = idx >> 7;
  const int sd = idx & 127;
  const int s  = sd >> 2;
  const int d  = sd & 3;
  dst[idx] = s2h[((size_t)s * NHID + n) * STACKD + d];
}

__global__ void init_state(unsigned* cnt, float* hbuf, f16_t* h16,
                           float* sbuf, float* svec) {
  const int idx = blockIdx.x * 256 + threadIdx.x;
  if (idx == 0) *cnt = 0u;
  if (idx < NHID) { hbuf[idx] = 0.f; h16[idx] = (f16_t)0.f; }
  if (idx < NSTACK * STACKD) sbuf[idx] = -1.f;
  if (idx < NSTACK * DEPTH) svec[idx] = -1.f;
}

// ---------------------------------------------------------------------------
// Persistent StackRNN kernel
// ---------------------------------------------------------------------------

__global__ void __launch_bounds__(256, 1)
stackrnn_persistent(const int* __restrict__ tokens,
                    const float* __restrict__ in2hid,
                    float* __restrict__ out,
                    unsigned* cnt,
                    float* hbuf, f16_t* h16buf,
                    float* sbuf, float* svec, float* logits,
                    const f16_t* __restrict__ Whh,
                    const f16_t* __restrict__ Wout,
                    const f16_t* __restrict__ Whead,
                    const float* __restrict__ Spack) {
  const int g    = blockIdx.x;
  const int tid  = threadIdx.x;
  const int lane = tid & 31;
  const int wid  = tid >> 5;

  __shared__ f16_t wlds[TILE_HALFS];   // 64KB weight tile (H and O WGs)
  __shared__ float red[128];
  __shared__ float out16[16];
  __shared__ float svec_s[128];
  __shared__ float headv[128];
  __shared__ float popw[NSTACK], pushw[NSTACK], noopw[NSTACK], topv[NSTACK];
  __shared__ float smx[256];

  // One-time staging of this WG's weight tile into LDS via async DMA.
  if (g < NWG_H + NWG_O) {
    const f16_t* src = (g < NWG_H) ? (Whh + (size_t)g * TILE_HALFS)
                                   : (Wout + (size_t)(g - NWG_H) * TILE_HALFS);
    stage_tile_async(src, wlds);
  }

  unsigned epoch = 0;

  for (int t = 0; t < T_STEPS; ++t) {
    const int par = t & 1;

    // ---------------- phase 1: h_t = sigmoid(...) ; pipelined softmax ------
    if (g < NWG_H) {
      if (tid < 128) svec_s[tid] = svec[tid];
      __syncthreads();
      wg_matvec16_lds(wlds, h16buf + (size_t)par * NHID, red, out16);
      // Stack-slice contribution: row r = tid>>4, 16 threads x 8 MACs each,
      // reduced with a shuffle tree inside each 16-lane group.
      const int r  = tid >> 4;
      const int cc = tid & 15;
      const float* sp = Spack + ((size_t)(g * 16 + r) * 128) + cc * 8;
      float acc = 0.f;
#pragma unroll
      for (int j = 0; j < 8; ++j) acc += sp[j] * svec_s[cc * 8 + j];
#pragma unroll
      for (int o = 8; o > 0; o >>= 1) acc += __shfl_xor(acc, o, 16);
      if (cc == 0) {
        const int row = g * 16 + r;
        const int x = tokens[t];
        float u = out16[r] + in2hid[(size_t)row * NCHAR + x] + acc;
        const float h = 1.f / (1.f + __expf(-u));
        hbuf[(size_t)(par ^ 1) * NHID + row]   = h;
        h16buf[(size_t)(par ^ 1) * NHID + row] = (f16_t)h;
      }
    } else if (g == WG_SMX && t > 0) {
      softmax256(logits, out + (size_t)(t - 1) * NCHAR, smx);
    }
    grid_sync(cnt, epoch);

    // ---------------- phase 2: heads + stack update ------------------------
    const f16_t* h16c = h16buf + (size_t)(par ^ 1) * NHID;
    if (g >= NWG_H && g < NWG_H + NWG_O) {
      const int ot = g - NWG_H;
      wg_matvec16_lds(wlds, h16c, red, out16);
      if (tid < 16) logits[ot * 16 + tid] = out16[tid];
    } else if (g == WG_ACT) {
      // Fused head matvec: wave `wid` owns row-tile `wid` (rows 16w..16w+15),
      // accumulating over all 64 k-tiles with no intermediate barriers.
      v8f c = {0.f, 0.f, 0.f, 0.f, 0.f, 0.f, 0.f, 0.f};
#pragma unroll 4
      for (int kt = 0; kt < 64; ++kt) {
        v16h a = *(const v16h*)(Whead + (((size_t)wid * 64 + kt) * 32 + lane) * 16);
        v16h b = *(const v16h*)(h16c + kt * 32 + ((lane >> 4) << 4));
        c = __builtin_amdgcn_wmma_f32_16x16x32_f16(false, a, false, b,
                                                   (short)0, c, false, false);
      }
      if (lane == 0 || lane == 16) {
        const int half = lane >> 4;
#pragma unroll
        for (int v = 0; v < 8; ++v) headv[wid * 16 + half * 8 + v] = c[v];
      }
      __syncthreads();
      if (tid < NSTACK) {
        const int s = tid;
        const float a0 = headv[s * 3 + 0];          // pop logit
        const float a1 = headv[s * 3 + 1];          // push logit
        const float a2 = headv[s * 3 + 2];          // noop logit
        const float m  = fmaxf(a0, fmaxf(a1, a2));
        const float e0 = __expf(a0 - m);
        const float e1 = __expf(a1 - m);
        const float e2 = __expf(a2 - m);
        const float inv = 1.f / (e0 + e1 + e2);
        popw[s]  = e0 * inv;
        pushw[s] = e1 * inv;
        noopw[s] = e2 * inv;
        float tv = headv[96 + s];
        tv = fminf(50.f, fmaxf(-50.f, tv));
        topv[s] = 1.f / (1.f + __expf(-tv));
      }
      __syncthreads();
      const float* sOld = sbuf + (size_t)par * (NSTACK * STACKD);
      float* sNew       = sbuf + (size_t)(par ^ 1) * (NSTACK * STACKD);
      for (int idx = tid; idx < NSTACK * STACKD; idx += 256) {
        const int s = idx >> 9;
        const int c2 = idx & (STACKD - 1);
        const float old    = sOld[idx];
        const float pushed = c2 ? sOld[idx - 1] : topv[s];
        const float popped = (c2 < STACKD - 1) ? sOld[idx + 1] : -1.f;
        const float nv = pushw[s] * pushed + popw[s] * popped + noopw[s] * old;
        sNew[idx] = nv;
        if (c2 < DEPTH) svec[s * DEPTH + c2] = nv;
      }
    }
    grid_sync(cnt, epoch);
  }

  // Epilogue: softmax for the final step's logits.
  if (g == WG_SMX) {
    softmax256(logits, out + (size_t)(T_STEPS - 1) * NCHAR, smx);
  }
}

// ---------------------------------------------------------------------------
// Launch
// ---------------------------------------------------------------------------

extern "C" void kernel_launch(void* const* d_in, const int* in_sizes, int n_in,
                              void* d_out, int out_size, void* d_ws, size_t ws_size,
                              hipStream_t stream) {
  const int*   tokens  = (const int*)d_in[0];
  const float* in2hid  = (const float*)d_in[1];   // [2048, 256]
  const float* hid2hid = (const float*)d_in[2];   // [2048, 2048]
  const float* hid2act = (const float*)d_in[3];   // [32, 3, 2048]
  const float* hid2stk = (const float*)d_in[4];   // [32, 512, 2048]
  const float* stk2hid = (const float*)d_in[5];   // [32, 2048, 512]
  const float* hid2out = (const float*)d_in[6];   // [256, 2048]
  float* out = (float*)d_out;                     // [4096, 256]

  char* ws = (char*)d_ws;
  size_t off = 0;
  auto alloc = [&](size_t bytes) -> char* {
    char* p = ws + off;
    off += (bytes + 255) & ~(size_t)255;
    return p;
  };

  unsigned* cnt   = (unsigned*)alloc(256);
  float*    hbuf  = (float*)alloc((size_t)2 * NHID * sizeof(float));
  f16_t*    h16   = (f16_t*)alloc((size_t)2 * NHID * sizeof(f16_t));
  float*    sbuf  = (float*)alloc((size_t)2 * NSTACK * STACKD * sizeof(float));
  float*    svec  = (float*)alloc((size_t)NSTACK * DEPTH * sizeof(float));
  float*    logit = (float*)alloc((size_t)NCHAR * sizeof(float));
  f16_t*    Whh   = (f16_t*)alloc((size_t)NHID * NHID * sizeof(f16_t));       // 8 MB
  f16_t*    Wout  = (f16_t*)alloc((size_t)NCHAR * NHID * sizeof(f16_t));      // 1 MB
  f16_t*    Whead = (f16_t*)alloc((size_t)128 * NHID * sizeof(f16_t));        // 0.5 MB
  float*    Spack = (float*)alloc((size_t)NHID * 128 * sizeof(float));        // 1 MB

  // Pre-pack weights into WMMA fragment layout (L2-resident afterwards).
  pack_wmma_tiles<<<(128 * TILE_HALFS) / 256, 256, 0, stream>>>(hid2hid, Whh, 128);
  pack_wmma_tiles<<<(16 * TILE_HALFS) / 256, 256, 0, stream>>>(hid2out, Wout, 16);
  pack_head_tiles<<<(8 * TILE_HALFS) / 256, 256, 0, stream>>>(hid2act, hid2stk, Whead);
  pack_spack<<<(NHID * 128) / 256, 256, 0, stream>>>(stk2hid, Spack);
  init_state<<<64, 256, 0, stream>>>(cnt, hbuf, h16, sbuf, svec);

  // One persistent kernel runs the whole sequential recurrence.
  stackrnn_persistent<<<NWG_TOT, 256, 0, stream>>>(
      tokens, in2hid, out, cnt, hbuf, h16, sbuf, svec, logit,
      Whh, Wout, Whead, Spack);
}
